// ArcMarginProductNonLinearSquarshing_73873437491541
// MI455X (gfx1250) — matrix operand order repaired and checked by reading, not compiled, not run on hardware
//
#include <hip/hip_runtime.h>
#include <hip/hip_bf16.h>
#include <math.h>

typedef __attribute__((ext_vector_type(16))) _Float16 v16h;
typedef __attribute__((ext_vector_type(8)))  _Float16 v8h;
typedef __attribute__((ext_vector_type(8)))  float    v8f;

#define BATCH   1024
#define KDIM    512
#define NCLS    50000
#define NTILE   256
#define NT_BLKS 196                 // ceil(50000/256)
#define CPAD    (NT_BLKS * NTILE)   // 50176 (zero padded classes)
#define MTILE   16

#define SC     30.0f
#define COS_M  0.8775825618903728f
#define SIN_M  0.479425538604203f
#define TH_C  (-0.8775825618903728f)   // cos(pi - 0.5)
#define MM_C   0.2397127693021015f     // sin(pi - 0.5) * 0.5

union AFrag { v16h v; v8h h[2]; };

// ---------------------------------------------------------------------------
// Kernel 1: x_scaled = x * sqrt(||x||^2) / (||x||^2 + 1)  ->  f16
// ---------------------------------------------------------------------------
__global__ __launch_bounds__(128) void prep_x_kernel(const float* __restrict__ x,
                                                     _Float16* __restrict__ Xh) {
  const int row = blockIdx.x;
  const int t   = threadIdx.x;            // 128 threads, 4 elems each
  const float* xr = x + (size_t)row * KDIM;
  float v[4]; float s = 0.f;
#pragma unroll
  for (int i = 0; i < 4; ++i) { v[i] = xr[t + 128 * i]; s += v[i] * v[i]; }
#pragma unroll
  for (int off = 16; off > 0; off >>= 1) s += __shfl_xor(s, off, 32);
  __shared__ float ls[4];
  if ((t & 31) == 0) ls[t >> 5] = s;
  __syncthreads();
  s = ls[0] + ls[1] + ls[2] + ls[3];
  const float scale = sqrtf(s) / (s + 1.0f);   // mul_s / ||x||
  _Float16* o = Xh + (size_t)row * KDIM;
#pragma unroll
  for (int i = 0; i < 4; ++i) o[t + 128 * i] = (_Float16)(v[i] * scale);
}

// ---------------------------------------------------------------------------
// Kernel 2: row-normalize weights -> f16, zero pad rows [NCLS, CPAD)
// ---------------------------------------------------------------------------
__global__ __launch_bounds__(128) void prep_w_kernel(const float* __restrict__ w,
                                                     _Float16* __restrict__ Wh) {
  const int row = blockIdx.x;             // 0..CPAD-1
  const int t   = threadIdx.x;
  _Float16* o = Wh + (size_t)row * KDIM;
  if (row >= NCLS) {
#pragma unroll
    for (int i = 0; i < 4; ++i) o[t + 128 * i] = (_Float16)0.0f;
    return;
  }
  const float* wr = w + (size_t)row * KDIM;
  float v[4]; float s = 0.f;
#pragma unroll
  for (int i = 0; i < 4; ++i) { v[i] = wr[t + 128 * i]; s += v[i] * v[i]; }
#pragma unroll
  for (int off = 16; off > 0; off >>= 1) s += __shfl_xor(s, off, 32);
  __shared__ float ls[4];
  if ((t & 31) == 0) ls[t >> 5] = s;
  __syncthreads();
  s = ls[0] + ls[1] + ls[2] + ls[3];
  const float scale = rsqrtf(s);
#pragma unroll
  for (int i = 0; i < 4; ++i) o[t + 128 * i] = (_Float16)(v[i] * scale);
}

// ---------------------------------------------------------------------------
// Kernel 3: 16x256 logit tile via v_wmma_f32_16x16x32_f16, fused
//           logsumexp / argmax(phi) partial reduction per class tile.
// partials[row * NT_BLKS + bx] = {max, sumexp, phimax, argmax-as-int}
// ---------------------------------------------------------------------------
__global__ __launch_bounds__(256) void gemm_tile_kernel(
    const _Float16* __restrict__ Xh, const _Float16* __restrict__ Wh,
    const int* __restrict__ label, float4* __restrict__ partials) {
  constexpr int PA = KDIM + 8;    // f16 pitch: conflict-free b128 LDS reads
  constexpr int PC = NTILE + 1;   // f32 pitch
  __shared__ _Float16 lA[MTILE * PA];
  __shared__ float    lC[MTILE * PC];

  const int bx    = blockIdx.x;          // class tile
  const int by    = blockIdx.y;          // batch tile
  const int tid   = threadIdx.x;
  const int lane  = tid & 31;
  const int wav   = tid >> 5;            // 8 waves
  const int Nbase = bx * NTILE;
  const int Mbase = by * MTILE;

  // --- cooperative A-tile load: 16 rows x 512 halves, as 16B chunks ---
  {
    const uint4* src = (const uint4*)(Xh + (size_t)Mbase * KDIM);
#pragma unroll
    for (int i = 0; i < 4; ++i) {
      const int ci = tid + 256 * i;          // 1024 chunks total
      const int r  = ci >> 6;                // 64 chunks per row
      const int c8 = ci & 63;
      *(uint4*)(&lA[r * PA + c8 * 8]) = src[ci];
    }
  }
  __syncthreads();

  // --- WMMA main loop: each wave owns 2 adjacent 16x16 N-subtiles ---
  v8f acc0 = {}; v8f acc1 = {};
  const int arow = lane & 15;
  const int asel = lane >> 4;                        // A: K 0-7/16-23 vs 8-15/24-31
  const int st0  = wav * 2, st1 = wav * 2 + 1;
  const int c0   = Nbase + st0 * 16 + (lane & 15);   // B column = class
  const int c1   = Nbase + st1 * 16 + (lane & 15);
  const int bko  = (lane >> 4) * 8;                  // B: K 0-7/16-23 vs 8-15/24-31
  const _Float16* w0 = Wh + (size_t)c0 * KDIM + bko;
  const _Float16* w1 = Wh + (size_t)c1 * KDIM + bko;

  for (int kb = 0; kb < KDIM; kb += 32) {
    if (kb + 32 < KDIM) {                            // global_prefetch_b8 hints
      __builtin_prefetch(w0 + kb + 32, 0, 0);
      __builtin_prefetch(w1 + kb + 32, 0, 0);
    }
    AFrag a, b0, b1;
    const _Float16* ap = &lA[arow * PA + kb + asel * 8];
    a.h[0]  = *(const v8h*)(ap);
    a.h[1]  = *(const v8h*)(ap + 16);
    b0.h[0] = *(const v8h*)(w0 + kb);
    b0.h[1] = *(const v8h*)(w0 + kb + 16);
    b1.h[0] = *(const v8h*)(w1 + kb);
    b1.h[1] = *(const v8h*)(w1 + kb + 16);
    acc0 = __builtin_amdgcn_wmma_f32_16x16x32_f16(false, a.v, false, b0.v,
                                                  (short)0, acc0, false, false);
    acc1 = __builtin_amdgcn_wmma_f32_16x16x32_f16(false, a.v, false, b1.v,
                                                  (short)0, acc1, false, false);
  }

  // --- spill cosines to LDS (C layout: VGPR r -> M = r + 8*(lane/16), N = lane%16)
#pragma unroll
  for (int r = 0; r < 8; ++r) {
    const int m = r + 8 * (lane >> 4);
    lC[m * PC + st0 * 16 + (lane & 15)] = acc0[r];
    lC[m * PC + st1 * 16 + (lane & 15)] = acc1[r];
  }
  __syncthreads();

  // --- fused reduction: wave w reduces rows 2w, 2w+1 over this class tile ---
#pragma unroll
  for (int rr = 0; rr < 2; ++rr) {
    const int row = wav * 2 + rr;
    const int Mg  = Mbase + row;
    const int lab = label[Mg];
    float m = -1e30f, ssum = 0.f, pm = -1e30f;
    int   pi = 0x7fffffff;
#pragma unroll
    for (int i = 0; i < 8; ++i) {
      const int n = lane + 32 * i;
      const int c = Nbase + n;
      if (c < NCLS) {
        const float cosv = lC[row * PC + n];
        const float sine = sqrtf(fmaxf(0.f, 1.f - cosv * cosv));
        const float phi  = (cosv > TH_C) ? (cosv * COS_M - sine * SIN_M)
                                         : (cosv - MM_C);
        const float ov = SC * ((c == lab) ? phi : cosv);
        if (ov <= m) { ssum += __expf(ov - m); }
        else         { ssum = ssum * __expf(m - ov) + 1.f; m = ov; }
        if (phi > pm || (phi == pm && c < pi)) { pm = phi; pi = c; }
      }
    }
    // wave32 merge
#pragma unroll
    for (int off = 16; off > 0; off >>= 1) {
      const float om  = __shfl_xor(m,    off, 32);
      const float os  = __shfl_xor(ssum, off, 32);
      const float opm = __shfl_xor(pm,   off, 32);
      const int   opi = __shfl_xor(pi,   off, 32);
      const float nm  = fmaxf(m, om);
      ssum = ssum * __expf(m - nm) + os * __expf(om - nm);
      m = nm;
      if (opm > pm || (opm == pm && opi < pi)) { pm = opm; pi = opi; }
    }
    if (lane == 0)
      partials[(size_t)Mg * NT_BLKS + bx] =
          make_float4(m, ssum, pm, __int_as_float(pi));
  }
}

// ---------------------------------------------------------------------------
// Kernel 4: per-row merge of NT_BLKS partials + label-logit + loss/acc flags
// ---------------------------------------------------------------------------
__global__ __launch_bounds__(256) void row_finalize_kernel(
    const float4* __restrict__ partials, const _Float16* __restrict__ Xh,
    const _Float16* __restrict__ Wh, const int* __restrict__ label,
    float* __restrict__ rowloss, float* __restrict__ rowcorr) {
  const int row = blockIdx.x;
  const int t   = threadIdx.x;
  const int lab = label[row];

  float m = -1e30f, s = 0.f, pm = -1e30f;
  int   pi = 0x7fffffff;
  if (t < NT_BLKS) {
    const float4 p = partials[(size_t)row * NT_BLKS + t];
    m = p.x; s = p.y; pm = p.z; pi = __float_as_int(p.w);
  }
  // label-class dot (same f16 data path as the GEMM)
  const _Float16* xr = Xh + (size_t)row * KDIM;
  const _Float16* wr = Wh + (size_t)lab * KDIM;
  float d = (float)xr[t] * (float)wr[t] + (float)xr[t + 256] * (float)wr[t + 256];

  __shared__ float sm[256], ss[256], sp[256], sd[256];
  __shared__ int   si[256];
  sm[t] = m; ss[t] = s; sp[t] = pm; si[t] = pi; sd[t] = d;
  __syncthreads();
  for (int off = 128; off > 0; off >>= 1) {
    if (t < off) {
      const float om = sm[t + off], os = ss[t + off], opm = sp[t + off];
      const int   opi = si[t + off];
      const float nm = fmaxf(sm[t], om);
      ss[t] = ss[t] * __expf(sm[t] - nm) + os * __expf(om - nm);
      sm[t] = nm;
      if (opm > sp[t] || (opm == sp[t] && opi < si[t])) { sp[t] = opm; si[t] = opi; }
      sd[t] += sd[t + off];
    }
    __syncthreads();
  }
  if (t == 0) {
    const float cosl = sd[0];
    const float sine = sqrtf(fmaxf(0.f, 1.f - cosl * cosl));
    const float phil = (cosl > TH_C) ? (cosl * COS_M - sine * SIN_M)
                                     : (cosl - MM_C);
    const float logz = sm[0] + __logf(ss[0]);
    rowloss[row] = logz - SC * phil;            // logz - output[row, label]
    rowcorr[row] = (si[0] == lab) ? 1.f : 0.f;  // argmax(phi) == label
  }
}

// ---------------------------------------------------------------------------
// Kernel 5: final means -> d_out = {loss, acc}
// ---------------------------------------------------------------------------
__global__ __launch_bounds__(256) void final_reduce_kernel(
    const float* __restrict__ rowloss, const float* __restrict__ rowcorr,
    float* __restrict__ out) {
  const int t = threadIdx.x;
  float l = 0.f, c = 0.f;
  for (int i = t; i < BATCH; i += 256) { l += rowloss[i]; c += rowcorr[i]; }
  __shared__ float sl[256], sc[256];
  sl[t] = l; sc[t] = c;
  __syncthreads();
  for (int off = 128; off > 0; off >>= 1) {
    if (t < off) { sl[t] += sl[t + off]; sc[t] += sc[t + off]; }
    __syncthreads();
  }
  if (t == 0) {
    out[0] = sl[0] / (float)BATCH;
    out[1] = sc[0] / (float)BATCH * 100.f;
  }
}

// ---------------------------------------------------------------------------
extern "C" void kernel_launch(void* const* d_in, const int* in_sizes, int n_in,
                              void* d_out, int out_size, void* d_ws, size_t ws_size,
                              hipStream_t stream) {
  (void)in_sizes; (void)n_in; (void)out_size; (void)ws_size;
  const float* x     = (const float*)d_in[0];   // [1024, 512]
  const int*   label = (const int*)d_in[1];     // [1024]
  const float* w     = (const float*)d_in[2];   // [50000, 512]

  char* p = (char*)d_ws;
  _Float16* Xh = (_Float16*)p;  p += (size_t)BATCH * KDIM * sizeof(_Float16);
  _Float16* Wh = (_Float16*)p;  p += (size_t)CPAD  * KDIM * sizeof(_Float16);
  float4* partials = (float4*)p; p += (size_t)BATCH * NT_BLKS * sizeof(float4);
  float* rowloss = (float*)p;    p += (size_t)BATCH * sizeof(float);
  float* rowcorr = (float*)p;    p += (size_t)BATCH * sizeof(float);

  prep_x_kernel<<<BATCH, 128, 0, stream>>>(x, Xh);
  prep_w_kernel<<<CPAD, 128, 0, stream>>>(w, Wh);
  gemm_tile_kernel<<<dim3(NT_BLKS, BATCH / MTILE), 256, 0, stream>>>(Xh, Wh, label, partials);
  row_finalize_kernel<<<BATCH, 256, 0, stream>>>(partials, Xh, Wh, label, rowloss, rowcorr);
  final_reduce_kernel<<<1, 256, 0, stream>>>(rowloss, rowcorr, (float*)d_out);
}
